// QJLAttention_80676665688762
// MI455X (gfx1250) — matrix operand order repaired and checked by reading, not compiled, not run on hardware
//
#include <hip/hip_runtime.h>
#include <hip/hip_bf16.h>
#include <math.h>

// ---------------- problem constants (match reference) ----------------
#define S_LEN 2048
#define HID_DIM 4096
#define NH 32
#define NKV 8
#define HD 128
#define SDIM 128

// ---------------- CDNA5 WMMA types ----------------
typedef __attribute__((ext_vector_type(16))) __bf16       v16bf;
typedef __attribute__((ext_vector_type(8)))  float        v8f;
typedef __attribute__((ext_vector_type(4)))  unsigned int u32x4;
typedef __attribute__((ext_vector_type(4)))  int          i32x4;

union FragU { u32x4 u[2]; v16bf v; };

// async global->LDS path (gfx1250); guarded so either toolchain compiles
#if defined(__has_builtin)
#if __has_builtin(__builtin_amdgcn_global_load_async_to_lds_b128)
#define HAVE_ASYNC_LDS 1
#endif
#if __has_builtin(__builtin_amdgcn_s_wait_asynccnt)
#define HAVE_WAIT_ASYNC 1
#endif
#endif

#ifdef HAVE_ASYNC_LDS
typedef __attribute__((address_space(1))) i32x4 gi32x4;   // global int4
typedef __attribute__((address_space(3))) i32x4 li32x4;   // LDS int4
#endif

__device__ __forceinline__ void wait_asynccnt0() {
#ifdef HAVE_WAIT_ASYNC
    __builtin_amdgcn_s_wait_asynccnt(0);
#else
    asm volatile("s_wait_asynccnt 0" ::: "memory");
#endif
}

__device__ __forceinline__ unsigned short f2bf(float f) {
    unsigned int u = __float_as_uint(f);
    u += 0x7FFFu + ((u >> 16) & 1u);   // round-to-nearest-even
    return (unsigned short)(u >> 16);
}

// A-fragment (16x32 bf16): lane holds 16 contiguous K values at +0..15 (elements)
__device__ __forceinline__ v16bf load_frag_a(const unsigned short* p) {
    FragU f;
    f.u[0] = *(const u32x4*)(p);
    f.u[1] = *(const u32x4*)(p + 8);
    return f.v;
}
// B-fragment when the 16 K-values are contiguous in memory (32 bytes)
__device__ __forceinline__ v16bf load_frag_contig(const unsigned short* p) {
    FragU f;
    f.u[0] = *(const u32x4*)(p);
    f.u[1] = *(const u32x4*)(p + 8);
    return f.v;
}

// ---------------- elementwise fp32 -> bf16 ----------------
__global__ void cvt_f32_bf16(const float* __restrict__ in,
                             unsigned short* __restrict__ out, int n) {
    int i = blockIdx.x * blockDim.x + threadIdx.x;
    if (i < n) out[i] = f2bf(in[i]);
}

// ---------------- pack row-major fp32 B[K,N] into WMMA B-fragment layout ---
// dst[((kt*ntiles+nt)*32 + lane)*16 + j] = bf16(B[(kt*32 + (lane>>4)*16 + j)*N + nt*16 + (lane&15)])
__global__ void pack_b(const float* __restrict__ B,
                       unsigned short* __restrict__ Bp, int K, int N) {
    int idx = blockIdx.x * blockDim.x + threadIdx.x;
    if (idx >= K * N) return;
    int j    = idx & 15;
    int lane = (idx >> 4) & 31;
    int tile = idx >> 9;
    int ntiles = N >> 4;
    int nt = tile % ntiles;
    int kt = tile / ntiles;
    int n = nt * 16 + (lane & 15);
    int k = kt * 32 + (lane >> 4) * 16 + j;
    Bp[idx] = f2bf(B[(size_t)k * N + n]);
}

// ---------------- transpose V [s][kvh*128+d] -> VT [kvh][d][s] ----------------
__global__ void transpose_v(const unsigned short* __restrict__ Vm,
                            unsigned short* __restrict__ VT, int n) {
    int i = blockIdx.x * blockDim.x + threadIdx.x;
    if (i >= n) return;
    int s = i >> 10;          // row stride NKV*HD = 1024
    int c = i & 1023;         // c = kvh*128 + d
    VT[(size_t)c * S_LEN + s] = Vm[i];
}

// ---------------- bf16 WMMA GEMM: 4-wave workgroup, 64x64 C macro-tile ------
// B slab for current K-step staged once per workgroup into LDS (async if
// available), shared by all 4 waves; each wave owns a 16-row sub-tile.
// MODE 0: fp32 out; MODE 1: bf16 out (x scale); MODE 2: bf16 sign(+1/-1)
template <int MODE>
__global__ void __launch_bounds__(128)
gemm16(const unsigned short* __restrict__ A,   // row-major [M,K] bf16
       const unsigned short* __restrict__ Bp,  // packed B-fragment layout
       void* __restrict__ C,
       int M, int N, int K, int lda, int ldc, float scale) {
    __shared__ __align__(16) unsigned short Bt[2048];   // 4 n-tiles x 512 bf16
    const int tid  = threadIdx.x;
    const int lane = tid & 31;
    const int wid  = tid >> 5;          // wave id -> M sub-tile
    const int mloc = lane & 15;
    const int hf   = lane >> 4;
    const int mt   = blockIdx.y * 4 + wid;
    const int nt0  = blockIdx.x << 2;
    const int ntiles = N >> 4;
    const unsigned short* Arow = A + (size_t)(mt * 16 + mloc) * lda;

    v8f acc0 = {}, acc1 = {}, acc2 = {}, acc3 = {};
    const int kt_n = K >> 5;
    for (int kt = 0; kt < kt_n; ++kt) {
        // ---- cooperative stage of the 4KB packed-B slab for this K-step ----
        const unsigned short* bsrc = Bp + (size_t)(kt * ntiles + nt0) * 512;
#ifdef HAVE_ASYNC_LDS
        __builtin_amdgcn_global_load_async_to_lds_b128(
            (gi32x4*)(bsrc + tid * 16), (li32x4*)(Bt + tid * 16), 0, 0);
        __builtin_amdgcn_global_load_async_to_lds_b128(
            (gi32x4*)(bsrc + tid * 16 + 8), (li32x4*)(Bt + tid * 16 + 8), 0, 0);
        wait_asynccnt0();
#else
        u32x4 t0 = *(const u32x4*)(bsrc + tid * 16);
        u32x4 t1 = *(const u32x4*)(bsrc + tid * 16 + 8);
        *(u32x4*)(Bt + tid * 16)     = t0;
        *(u32x4*)(Bt + tid * 16 + 8) = t1;
#endif
        __syncthreads();

        __builtin_prefetch(Arow + kt * 32 + 256, 0, 0);   // global_prefetch_b8
        v16bf a  = load_frag_a(Arow + kt * 32 + hf * 16);
        v16bf b0 = load_frag_contig(&Bt[lane * 16]);
        v16bf b1 = load_frag_contig(&Bt[512  + lane * 16]);
        v16bf b2 = load_frag_contig(&Bt[1024 + lane * 16]);
        v16bf b3 = load_frag_contig(&Bt[1536 + lane * 16]);
        acc0 = __builtin_amdgcn_wmma_f32_16x16x32_bf16(false, a, false, b0, (short)0, acc0, false, false);
        acc1 = __builtin_amdgcn_wmma_f32_16x16x32_bf16(false, a, false, b1, (short)0, acc1, false, false);
        acc2 = __builtin_amdgcn_wmma_f32_16x16x32_bf16(false, a, false, b2, (short)0, acc2, false, false);
        acc3 = __builtin_amdgcn_wmma_f32_16x16x32_bf16(false, a, false, b3, (short)0, acc3, false, false);
        __syncthreads();   // protect Bt before next K-step restage
    }
    v8f accs[4] = {acc0, acc1, acc2, acc3};
    for (int j = 0; j < 4; ++j) {
        int n = (nt0 + j) * 16 + mloc;
        for (int r = 0; r < 8; ++r) {
            int row = mt * 16 + hf * 8 + r;   // C layout: m = r + 8*(lane>>4)
            float v = accs[j][r] * scale;
            if (MODE == 0)
                ((float*)C)[(size_t)row * ldc + n] = v;
            else if (MODE == 1)
                ((unsigned short*)C)[(size_t)row * ldc + n] = f2bf(v);
            else
                ((unsigned short*)C)[(size_t)row * ldc + n] =
                    (v > 0.0f) ? (unsigned short)0x3F80u : (unsigned short)0xBF80u;
        }
    }
}

// ---------------- flash-style QJL attention, one wave per (16 queries, head) --
// QP: [s][H][SD] bf16 (JL*scale pre-applied), KS: [s][HKV][SD] bf16 (+-1),
// VT: [kvh][d][s] bf16, Oout: [s][H][D] bf16
__global__ void __launch_bounds__(32)
qjl_attn(const unsigned short* __restrict__ QP,
         const unsigned short* __restrict__ KS,
         const unsigned short* __restrict__ VT,
         unsigned short* __restrict__ Oout) {
    __shared__ __align__(16) unsigned short Pst[16 * 32];   // P tile staging
    const int lane = threadIdx.x & 31;
    const int mloc = lane & 15;
    const int hf   = lane >> 4;
    const int qb   = blockIdx.x;     // 16-query block, 0..127
    const int h    = blockIdx.y;     // 0..31
    const int kvh  = h >> 2;         // GQA: 4 reps per kv head

    // Preload 16x128 QP tile as 4 A-fragments (kept in VGPRs for whole k loop)
    v16bf qf[4];
    const unsigned short* qrow = QP + ((size_t)(qb * 16 + mloc) * NH + h) * SDIM;
    for (int et = 0; et < 4; ++et)
        qf[et] = load_frag_a(qrow + et * 32 + hf * 16);

    v8f Oacc[8] = {};                // 16 x 128 output accumulator (fp32)
    float mrow[8], lrow[8];
    for (int r = 0; r < 8; ++r) { mrow[r] = -__builtin_inff(); lrow[r] = 0.0f; }

    const int nkb   = qb + 1;        // valid 16-key blocks (causal)
    const int nkb32 = (nkb + 1) >> 1;

    for (int kb32 = 0; kb32 < nkb32; ++kb32) {
        float s0[8], s1[8];
        // ---- score subtile 0 (always valid) ----
        {
            const int kb = 2 * kb32;
            v8f c = {};
            for (int et = 0; et < 4; ++et) {
                const unsigned short* bp =
                    KS + ((size_t)(kb * 16 + mloc) * NKV + kvh) * SDIM + et * 32 + hf * 16;
                v16bf b = load_frag_contig(bp);
                c = __builtin_amdgcn_wmma_f32_16x16x32_bf16(false, qf[et], false, b, (short)0, c, false, false);
            }
            const bool diag = (kb == qb);
            for (int r = 0; r < 8; ++r) {
                float v = c[r];
                if (diag && (mloc > r + 8 * hf)) v = -__builtin_inff();
                s0[r] = v;
            }
        }
        // ---- score subtile 1 (may be beyond causal limit) ----
        const int kb1 = 2 * kb32 + 1;
        if (kb1 < nkb) {
            v8f c = {};
            for (int et = 0; et < 4; ++et) {
                const unsigned short* bp =
                    KS + ((size_t)(kb1 * 16 + mloc) * NKV + kvh) * SDIM + et * 32 + hf * 16;
                v16bf b = load_frag_contig(bp);
                c = __builtin_amdgcn_wmma_f32_16x16x32_bf16(false, qf[et], false, b, (short)0, c, false, false);
            }
            const bool diag = (kb1 == qb);
            for (int r = 0; r < 8; ++r) {
                float v = c[r];
                if (diag && (mloc > r + 8 * hf)) v = -__builtin_inff();
                s1[r] = v;
            }
        } else {
            for (int r = 0; r < 8; ++r) s1[r] = -__builtin_inff();
        }

        // ---- online softmax over the 32 new columns ----
        for (int r = 0; r < 8; ++r) {
            float mx = fmaxf(s0[r], s1[r]);
            for (int off = 1; off < 16; off <<= 1)     // reduce within 16-lane half
                mx = fmaxf(mx, __shfl_xor(mx, off, 32));
            float newm  = fmaxf(mrow[r], mx);
            float alpha = __expf(mrow[r] - newm);
            float p0 = __expf(s0[r] - newm);
            float p1 = __expf(s1[r] - newm);           // -inf -> 0 for masked cols
            float sm = p0 + p1;
            for (int off = 1; off < 16; off <<= 1)
                sm += __shfl_xor(sm, off, 32);
            lrow[r] = lrow[r] * alpha + sm;
            mrow[r] = newm;
            for (int db = 0; db < 8; ++db) Oacc[db][r] *= alpha;
            const int mm = r + 8 * hf;                 // stage P into LDS (16x32)
            Pst[mm * 32 + mloc]      = f2bf(p0);
            Pst[mm * 32 + 16 + mloc] = f2bf(p1);
        }
        __syncthreads();
        v16bf pf = load_frag_a(&Pst[mloc * 32 + hf * 16]);
        __syncthreads();

        // ---- O += P (16x32) @ V (32x128), 8 column chunks ----
        for (int db = 0; db < 8; ++db) {
            const unsigned short* vp =
                VT + ((size_t)(kvh * HD + db * 16 + mloc)) * S_LEN + kb32 * 32 + hf * 16;
            v16bf vb = load_frag_contig(vp);
            Oacc[db] = __builtin_amdgcn_wmma_f32_16x16x32_bf16(false, pf, false, vb, (short)0, Oacc[db], false, false);
        }
    }

    // ---- normalize and store bf16 [s][h][d] ----
    for (int db = 0; db < 8; ++db) {
        for (int r = 0; r < 8; ++r) {
            int mm = r + 8 * hf;
            int d  = db * 16 + mloc;
            float o = Oacc[db][r] / lrow[r];
            Oout[((size_t)(qb * 16 + mm) * NH + h) * HD + d] = f2bf(o);
        }
    }
}

// ---------------- host-side orchestration ----------------
extern "C" void kernel_launch(void* const* d_in, const int* in_sizes, int n_in,
                              void* d_out, int out_size, void* d_ws, size_t ws_size,
                              hipStream_t stream) {
    const float* x  = (const float*)d_in[0];   // [2048,4096]
    const float* Wq = (const float*)d_in[1];   // [4096,4096]
    const float* Wk = (const float*)d_in[2];   // [4096,1024]
    const float* Wv = (const float*)d_in[3];   // [4096,1024]
    const float* Wo = (const float*)d_in[4];   // [4096,4096]
    const float* G  = (const float*)d_in[5];   // [128,128]
    float* out = (float*)d_out;

    char* ws = (char*)d_ws;
    size_t off = 0;
    auto alloc = [&](size_t bytes) -> unsigned short* {
        unsigned short* p = (unsigned short*)(ws + off);
        off += (bytes + 255) & ~(size_t)255;
        return p;
    };

    const size_t nX  = (size_t)S_LEN * HID_DIM;       // 8388608
    const size_t nKV = (size_t)S_LEN * NKV * HD;      // 2097152
    unsigned short* xb  = alloc(nX * 2);
    unsigned short* Wqp = alloc((size_t)HID_DIM * HID_DIM * 2);
    unsigned short* Wkp = alloc((size_t)HID_DIM * NKV * HD * 2);
    unsigned short* Wvp = alloc((size_t)HID_DIM * NKV * HD * 2);
    unsigned short* Wop = alloc((size_t)HID_DIM * HID_DIM * 2);
    unsigned short* Gp  = alloc((size_t)HD * SDIM * 2);
    unsigned short* Qb  = alloc(nX * 2);
    unsigned short* Kb  = alloc(nKV * 2);
    unsigned short* Vb  = alloc(nKV * 2);
    unsigned short* VTb = alloc(nKV * 2);
    unsigned short* QPb = alloc(nX * 2);
    unsigned short* KSb = alloc(nKV * 2);
    unsigned short* AOb = alloc(nX * 2);
    (void)ws_size; (void)in_sizes; (void)n_in; (void)out_size;

    const int T = 256;
    cvt_f32_bf16<<<(int)((nX + T - 1) / T), T, 0, stream>>>(x, xb, (int)nX);
    pack_b<<<(int)((HID_DIM * (size_t)HID_DIM + T - 1) / T), T, 0, stream>>>(Wq, Wqp, HID_DIM, HID_DIM);
    pack_b<<<(int)((HID_DIM * (size_t)(NKV * HD) + T - 1) / T), T, 0, stream>>>(Wk, Wkp, HID_DIM, NKV * HD);
    pack_b<<<(int)((HID_DIM * (size_t)(NKV * HD) + T - 1) / T), T, 0, stream>>>(Wv, Wvp, HID_DIM, NKV * HD);
    pack_b<<<(int)((HID_DIM * (size_t)HID_DIM + T - 1) / T), T, 0, stream>>>(Wo, Wop, HID_DIM, HID_DIM);
    pack_b<<<(int)((HD * SDIM + T - 1) / T), T, 0, stream>>>(G, Gp, HD, SDIM);

    // Q/K/V projections (bf16 out); 4-wave blocks, grid.y = M/64
    gemm16<1><<<dim3(HID_DIM / 64, S_LEN / 64), 128, 0, stream>>>(
        xb, Wqp, Qb, S_LEN, HID_DIM, HID_DIM, HID_DIM, HID_DIM, 1.0f);
    gemm16<1><<<dim3((NKV * HD) / 64, S_LEN / 64), 128, 0, stream>>>(
        xb, Wkp, Kb, S_LEN, NKV * HD, HID_DIM, HID_DIM, NKV * HD, 1.0f);
    gemm16<1><<<dim3((NKV * HD) / 64, S_LEN / 64), 128, 0, stream>>>(
        xb, Wvp, Vb, S_LEN, NKV * HD, HID_DIM, HID_DIM, NKV * HD, 1.0f);
    transpose_v<<<(int)((nKV + T - 1) / T), T, 0, stream>>>(Vb, VTb, (int)nKV);

    // QJL projections: fold sqrt(pi/2)/sqrt(D) into QP; k_sign via sign epilogue
    const float qscale = 0.11077818f;   // sqrt(pi/2) / sqrt(128)
    gemm16<1><<<dim3(SDIM / 64, (S_LEN * NH) / 64), 128, 0, stream>>>(
        Qb, Gp, QPb, S_LEN * NH, SDIM, HD, HD, SDIM, qscale);
    gemm16<2><<<dim3(SDIM / 64, (S_LEN * NKV) / 64), 128, 0, stream>>>(
        Kb, Gp, KSb, S_LEN * NKV, SDIM, HD, HD, SDIM, 1.0f);

    // Flash-style causal attention
    qjl_attn<<<dim3(S_LEN / 16, NH), 32, 0, stream>>>(QPb, KSb, VTb, AOb);

    // Output projection (fp32 out)
    gemm16<0><<<dim3(HID_DIM / 64, S_LEN / 64), 128, 0, stream>>>(
        AOb, Wop, out, S_LEN, HID_DIM, HID_DIM, HID_DIM, HID_DIM, 1.0f);
}
